// MultiHead_Attention_68848325755622
// MI455X (gfx1250) — matrix operand order repaired and checked by reading, not compiled, not run on hardware
//
#include <hip/hip_runtime.h>
#include <cstdint>

#define DEVINL __device__ __forceinline__

typedef __bf16 bf16;
typedef bf16  v8bf  __attribute__((ext_vector_type(8)));
typedef bf16  v16bf __attribute__((ext_vector_type(16)));
typedef float v8f   __attribute__((ext_vector_type(8)));
typedef uint32_t v4u __attribute__((ext_vector_type(4)));
typedef int      v8i __attribute__((ext_vector_type(8)));
typedef int      v4i __attribute__((ext_vector_type(4)));

constexpr int Bsz = 4, Tsz = 2048, Csz = 1024, Hn = 16, DKc = 64;
constexpr int Msz = Bsz * Tsz;       // 8192 rows in all GEMMs
constexpr int MW  = Tsz / 32;        // mask words per row

#if defined(__has_builtin)
#if __has_builtin(__builtin_amdgcn_tensor_load_to_lds) && \
    __has_builtin(__builtin_amdgcn_s_wait_tensorcnt)
#define HAVE_TDM 1
#endif
#endif

// ---------------------------------------------------------------------------
// WMMA fragment loaders (CDNA5 16-bit layouts, CDNA5 ISA §7.12.2)
// A (16x32, MxK): lanes 0-15 -> row M=lane, K = 0..7 then 16..23
//                 lanes 16-31 -> row M=lane-16, K = 8..15 then 24..31
// B (32x16, KxN): lane holds column N=lane&15; lanes 0-15 K=0..7 then 16..23,
//                 lanes 16-31 K=8..15 then 24..31 (same as A)
// C/D (16x16 f32): elem e of v8f -> M = e + 8*(lane>=16), N = lane&15
// ---------------------------------------------------------------------------
DEVINL v16bf frag_a(const bf16* base, int stride, int lane) {
  int row  = lane & 15;
  int koff = (lane >> 4) << 3;              // 0 or 8
  const bf16* p = base + (size_t)row * stride + koff;
  v8bf lo = *(const v8bf*)p;
  v8bf hi = *(const v8bf*)(p + 16);
  v16bf r;
#pragma unroll
  for (int i = 0; i < 8; ++i) { r[i] = lo[i]; r[i + 8] = hi[i]; }
  return r;
}

DEVINL v16bf frag_b(const bf16* base, int stride, int lane) {
  int col  = lane & 15;
  int koff = (lane >> 4) << 3;              // 0 or 8
  const bf16* p = base + (size_t)col * stride + koff;
  v8bf lo = *(const v8bf*)p;
  v8bf hi = *(const v8bf*)(p + 16);
  v16bf r;
#pragma unroll
  for (int i = 0; i < 8; ++i) { r[i] = lo[i]; r[i + 8] = hi[i]; }
  return r;
}

DEVINL v8f wmma_bf16(v16bf a, v16bf b, v8f c) {
  return __builtin_amdgcn_wmma_f32_16x16x32_bf16(
      false, a, false, b, (short)0, c, false, false);
}

#if HAVE_TDM
// 2D tensor_load_to_lds: tile (tile0 x tile1 elements, 2B each) from a
// row-major tensor with row stride stride0, into contiguous LDS at lds_off.
// D# packing per CDNA5 ISA.
DEVINL void tdm_load_2d(const void* gaddr, uint32_t lds_off,
                        uint32_t td0, uint32_t td1,
                        uint32_t tile0, uint32_t tile1, uint64_t stride0) {
  uint64_t ga = (uint64_t)(uintptr_t)gaddr;
  v4u g0;
  g0[0] = 1u;                                   // count=1, user descriptor
  g0[1] = lds_off;                              // lds_addr (bytes)
  g0[2] = (uint32_t)ga;                         // global_addr[31:0]
  g0[3] = (uint32_t)((ga >> 32) & 0x01FFFFFFu)  // global_addr[56:32]
        | (2u << 30);                           // type=2 (image)
  v8i g1;
  g1[0] = (int)(1u << 16);                      // data_size=1 -> 2 bytes
  g1[1] = (int)((td0 & 0xFFFFu) << 16);         // tensor_dim0[15:0]
  g1[2] = (int)((td0 >> 16) | ((td1 & 0xFFFFu) << 16));
  g1[3] = (int)((td1 >> 16) | (tile0 << 16));   // tile_dim0
  g1[4] = (int)tile1;                           // tile_dim1 (tile_dim2=0)
  g1[5] = (int)(uint32_t)stride0;               // tensor_dim0_stride[31:0]
  g1[6] = (int)(uint32_t)(stride0 >> 32);       // stride[47:32], dim1_stride=0
  g1[7] = 0;
  v4i z4 = {0, 0, 0, 0};
#if __clang_major__ >= 23
  v8i z8 = {0, 0, 0, 0, 0, 0, 0, 0};
  __builtin_amdgcn_tensor_load_to_lds(g0, g1, z4, z4, z8, 0);
#else
  __builtin_amdgcn_tensor_load_to_lds(g0, g1, z4, z4, 0);
#endif
}
DEVINL uint32_t lds_off(const void* p) { return (uint32_t)(uintptr_t)p; }
#endif

// ---------------------------------------------------------------------------
// fp32 -> bf16 conversion, 8 elements / thread
// ---------------------------------------------------------------------------
__global__ __launch_bounds__(256)
void f2bf8(const float* __restrict__ in, bf16* __restrict__ out, int n) {
  int i = (blockIdx.x * blockDim.x + threadIdx.x) * 8;
  if (i + 7 < n) {
    v8bf o;
#pragma unroll
    for (int j = 0; j < 8; ++j) o[j] = (bf16)in[i + j];
    *(v8bf*)(out + i) = o;
  }
}

// ---------------------------------------------------------------------------
// Bit-pack mask: one wave32 ballot per 32 ints -> [B,T,T/32] words.
// Cuts mask read traffic 32x (1.07 GB -> 33 MB across all heads).
// ---------------------------------------------------------------------------
__global__ __launch_bounds__(256)
void pack_mask(const int* __restrict__ m, unsigned* __restrict__ bits) {
  size_t i = (size_t)blockIdx.x * 256 + threadIdx.x;
  unsigned long long bal = __ballot(m[i] != 0);
  if ((threadIdx.x & 31) == 0) bits[i >> 5] = (unsigned)bal;
}

// ---------------------------------------------------------------------------
// GEMM: out[m,n] = sum_k A[m,k] * W[n,k] + bias[n]
// mode 0: bf16 [B,H,T,DK]; mode 1: bf16 [B,H,DK,T]; mode 2: fp32 [M,C]
// ---------------------------------------------------------------------------
__global__ __launch_bounds__(256)
void gemm_wmma(const bf16* __restrict__ A, const bf16* __restrict__ W,
               const float* __restrict__ bias, void* __restrict__ out, int mode) {
  const int lane = threadIdx.x & 31;
  const int wave = threadIdx.x >> 5;
  const int m0 = blockIdx.y * 128 + (wave >> 1) * 32;
  const int n0 = blockIdx.x * 128 + (wave & 1) * 64;

  v8f acc[2][4];
#pragma unroll
  for (int i = 0; i < 2; ++i)
#pragma unroll
    for (int j = 0; j < 4; ++j) acc[i][j] = {};

  for (int k0 = 0; k0 < Csz; k0 += 32) {
    v16bf af[2], bw[4];
#pragma unroll
    for (int i = 0; i < 2; ++i)
      af[i] = frag_a(A + (size_t)(m0 + 16 * i) * Csz + k0, Csz, lane);
#pragma unroll
    for (int j = 0; j < 4; ++j)
      bw[j] = frag_b(W + (size_t)(n0 + 16 * j) * Csz + k0, Csz, lane);
#pragma unroll
    for (int i = 0; i < 2; ++i)
#pragma unroll
      for (int j = 0; j < 4; ++j)
        acc[i][j] = wmma_bf16(af[i], bw[j], acc[i][j]);
  }

  const int half = lane >> 4, col = lane & 15;
#pragma unroll
  for (int i = 0; i < 2; ++i)
#pragma unroll
    for (int j = 0; j < 4; ++j)
#pragma unroll
      for (int e = 0; e < 8; ++e) {
        int m = m0 + 16 * i + e + 8 * half;
        int n = n0 + 16 * j + col;
        float val = acc[i][j][e] + bias[n];
        int b = m >> 11, t = m & (Tsz - 1);
        int h = n >> 6,  d = n & 63;
        if (mode == 0) {
          ((bf16*)out)[((size_t)(b * Hn + h) * Tsz + t) * DKc + d] = (bf16)val;
        } else if (mode == 1) {
          ((bf16*)out)[((size_t)(b * Hn + h) * DKc + d) * Tsz + t] = (bf16)val;
        } else {
          ((float*)out)[(size_t)m * Csz + n] = val;
        }
      }
}

// ---------------------------------------------------------------------------
// Flash-style masked attention. grid = (T/128, B*H); 8 waves/block.
// K/V tiles staged once per workgroup in LDS (TDM double-buffered when
// available) instead of 8x redundant per-wave global reads.
// ---------------------------------------------------------------------------
__global__ __launch_bounds__(256)
void attn_wmma(const bf16* __restrict__ Q, const bf16* __restrict__ K,
               const bf16* __restrict__ V, const unsigned* __restrict__ mbits,
               bf16* __restrict__ Y) {
  __shared__ bf16 Kl[2][32 * 64];            // key-block tiles   (2 x 4 KB)
  __shared__ bf16 Vl[2][64 * 32];            // value-block tiles (2 x 4 KB)
  __shared__ bf16 plds[8][16 * 32];          // per-wave P staging (8 KB)

  const int lane = threadIdx.x & 31;
  const int wave = threadIdx.x >> 5;
  const int half = lane >> 4, col = lane & 15;
  const int bh = blockIdx.y;
  const int b = bh >> 4, h = bh & 15;
  const int q0 = blockIdx.x * 128 + wave * 16;

  const bf16* Qb = Q + ((size_t)bh * Tsz + q0) * DKc;
  const bf16* Kb = K + (size_t)bh * Tsz * DKc;
  const bf16* Vb = V + (size_t)bh * DKc * Tsz;
  const size_t rowbase = (size_t)(b * Tsz + q0 + 8 * half) * MW;

  v16bf aq[2];
#pragma unroll
  for (int kk = 0; kk < 2; ++kk) aq[kk] = frag_a(Qb + kk * 32, DKc, lane);

  float m_i[8], l_i[8];
  v8f o[4];
#pragma unroll
  for (int e = 0; e < 8; ++e) { m_i[e] = -3.0e38f; l_i[e] = 0.f; }
#pragma unroll
  for (int d = 0; d < 4; ++d) o[d] = {};

  const float scale = 0.125f;                // 1/sqrt(DK)

  auto kv_block = [&](const bf16* Klp, const bf16* Vlp, int j0) {
    // S = Q @ K^T for a 16x32 score tile (frags from LDS)
    v8f s[2]; s[0] = {}; s[1] = {};
#pragma unroll
    for (int j = 0; j < 2; ++j)
#pragma unroll
      for (int kk = 0; kk < 2; ++kk) {
        v16bf bk = frag_b(Klp + (16 * j) * DKc + kk * 32, DKc, lane);
        s[j] = wmma_bf16(aq[kk], bk, s[j]);
      }
    // mask from packed bits: one broadcast word per query row
    unsigned mw[8];
#pragma unroll
    for (int e = 0; e < 8; ++e)
      mw[e] = mbits[rowbase + (size_t)e * MW + (j0 >> 5)];
#pragma unroll
    for (int j = 0; j < 2; ++j)
#pragma unroll
      for (int e = 0; e < 8; ++e) {
        float v = s[j][e] * scale;
        s[j][e] = ((mw[e] >> (j * 16 + col)) & 1u) ? v : -1.0e9f;
      }
    // row-wise max across the 16 lanes of each half (C-layout rows)
    float rmax[8];
#pragma unroll
    for (int e = 0; e < 8; ++e) rmax[e] = fmaxf(s[0][e], s[1][e]);
#pragma unroll
    for (int x = 1; x < 16; x <<= 1)
#pragma unroll
      for (int e = 0; e < 8; ++e)
        rmax[e] = fmaxf(rmax[e], __shfl_xor(rmax[e], x, 32));
    // online-softmax update
    float rsum[8];
#pragma unroll
    for (int e = 0; e < 8; ++e) {
      float mn = fmaxf(m_i[e], rmax[e]);
      float corr = __expf(m_i[e] - mn);
      m_i[e] = mn;
      l_i[e] *= corr;
#pragma unroll
      for (int d = 0; d < 4; ++d) o[d][e] *= corr;
      rsum[e] = 0.f;
    }
#pragma unroll
    for (int j = 0; j < 2; ++j)
#pragma unroll
      for (int e = 0; e < 8; ++e) {
        float p = __expf(s[j][e] - m_i[e]);
        s[j][e] = p;
        rsum[e] += p;
      }
#pragma unroll
    for (int x = 1; x < 16; x <<= 1)
#pragma unroll
      for (int e = 0; e < 8; ++e) rsum[e] += __shfl_xor(rsum[e], x, 32);
#pragma unroll
    for (int e = 0; e < 8; ++e) l_i[e] += rsum[e];

    // P: C-layout -> A-layout via wave-private LDS round trip
    bf16* pl = &plds[wave][0];
#pragma unroll
    for (int j = 0; j < 2; ++j)
#pragma unroll
      for (int e = 0; e < 8; ++e)
        pl[(e + 8 * half) * 32 + j * 16 + col] = (bf16)s[j][e];
    asm volatile("" ::: "memory");
    v16bf pf = frag_a(pl, 32, lane);
    asm volatile("" ::: "memory");

    // O += P @ V (V tile [64dk][32key] in LDS, keys contiguous)
#pragma unroll
    for (int d = 0; d < 4; ++d) {
      v16bf vf = frag_b(Vlp + (16 * d) * 32, 32, lane);
      o[d] = wmma_bf16(pf, vf, o[d]);
    }
  };

#if HAVE_TDM
  // Double-buffered TDM pipeline: wave 0 drives the Tensor Data Mover,
  // 2 descriptors (K tile 32x64, V tile 64x32) per key block.
  if (wave == 0) {
    tdm_load_2d(Kb, lds_off(&Kl[0][0]), DKc, Tsz, DKc, 32, DKc);
    tdm_load_2d(Vb, lds_off(&Vl[0][0]), Tsz, DKc, 32, DKc, Tsz);
  }
  for (int j0 = 0, it = 0; j0 < Tsz; j0 += 32, ++it) {
    int cur = it & 1;
    if (wave == 0) {
      if (j0 + 32 < Tsz) {
        tdm_load_2d(Kb + (size_t)(j0 + 32) * DKc, lds_off(&Kl[cur ^ 1][0]),
                    DKc, Tsz, DKc, 32, DKc);
        tdm_load_2d(Vb + (j0 + 32), lds_off(&Vl[cur ^ 1][0]),
                    Tsz, DKc, 32, DKc, Tsz);
        __builtin_amdgcn_s_wait_tensorcnt(2);  // current pair complete
      } else {
        __builtin_amdgcn_s_wait_tensorcnt(0);
      }
    }
    __syncthreads();                           // publish LDS tile
    kv_block(&Kl[cur][0], &Vl[cur][0], j0);
    __syncthreads();                           // reads done before re-issue
  }
#else
  // Fallback: cooperative 256-thread copy (each thread moves 16B of K and V)
  const int t = threadIdx.x;
  for (int j0 = 0; j0 < Tsz; j0 += 32) {
    *(v8bf*)(&Kl[0][0] + t * 8) =
        *(const v8bf*)(Kb + (size_t)(j0 + (t >> 3)) * DKc + (t & 7) * 8);
    *(v8bf*)(&Vl[0][0] + t * 8) =
        *(const v8bf*)(Vb + (size_t)(t >> 2) * Tsz + j0 + (t & 3) * 8);
    __syncthreads();
    kv_block(&Kl[0][0], &Vl[0][0], j0);
    __syncthreads();
  }
#endif

  // normalize and store Y as [B,T,H,DK] bf16
#pragma unroll
  for (int e = 0; e < 8; ++e) {
    int q = q0 + e + 8 * half;
    float inv = 1.f / l_i[e];
#pragma unroll
    for (int d = 0; d < 4; ++d) {
      int dd = d * 16 + col;
      Y[(((size_t)b * Tsz + q) * Hn + h) * DKc + dd] = (bf16)(o[d][e] * inv);
    }
  }
}

// ---------------------------------------------------------------------------
extern "C" void kernel_launch(void* const* d_in, const int* in_sizes, int n_in,
                              void* d_out, int out_size, void* d_ws, size_t ws_size,
                              hipStream_t stream) {
  const float* q   = (const float*)d_in[0];
  const float* k   = (const float*)d_in[1];
  const float* v   = (const float*)d_in[2];
  const int*   msk = (const int*)d_in[3];
  const float* Wq  = (const float*)d_in[4];
  const float* bq  = (const float*)d_in[5];
  const float* Wk  = (const float*)d_in[6];
  const float* bk  = (const float*)d_in[7];
  const float* Wv  = (const float*)d_in[8];
  const float* bv  = (const float*)d_in[9];
  const float* Wf  = (const float*)d_in[10];
  const float* bf_ = (const float*)d_in[11];

  char* ws = (char*)d_ws;
  size_t off = 0;
  auto alloc = [&](size_t bytes) {
    char* p = ws + off;
    off += (bytes + 255) & ~(size_t)255;
    return p;
  };
  const size_t actN = (size_t)Msz * Csz;     // 8 Mi elements
  const size_t wN   = (size_t)Csz * Csz;     // 1 Mi elements
  bf16* xq  = (bf16*)alloc(actN * 2);
  bf16* xk  = (bf16*)alloc(actN * 2);
  bf16* xv  = (bf16*)alloc(actN * 2);
  bf16* wqb = (bf16*)alloc(wN * 2);
  bf16* wkb = (bf16*)alloc(wN * 2);
  bf16* wvb = (bf16*)alloc(wN * 2);
  bf16* wfb = (bf16*)alloc(wN * 2);
  bf16* Qh  = (bf16*)alloc(actN * 2);        // [B,H,T,DK]
  bf16* Kh  = (bf16*)alloc(actN * 2);        // [B,H,T,DK]
  bf16* Vh  = (bf16*)alloc(actN * 2);        // [B,H,DK,T]
  bf16* Yb  = (bf16*)alloc(actN * 2);        // [B,T,C]
  unsigned* mb = (unsigned*)alloc(((size_t)Bsz * Tsz * MW) * 4);  // packed mask

  const int cBlkA = (int)(actN / 8 / 256);   // 4096
  const int cBlkW = (int)(wN / 8 / 256);     // 512
  f2bf8<<<cBlkA, 256, 0, stream>>>(q,  xq, (int)actN);
  f2bf8<<<cBlkA, 256, 0, stream>>>(k,  xk, (int)actN);
  f2bf8<<<cBlkA, 256, 0, stream>>>(v,  xv, (int)actN);
  f2bf8<<<cBlkW, 256, 0, stream>>>(Wq, wqb, (int)wN);
  f2bf8<<<cBlkW, 256, 0, stream>>>(Wk, wkb, (int)wN);
  f2bf8<<<cBlkW, 256, 0, stream>>>(Wv, wvb, (int)wN);
  f2bf8<<<cBlkW, 256, 0, stream>>>(Wf, wfb, (int)wN);

  const int mBlk = (int)((size_t)Bsz * Tsz * Tsz / 256);  // 65536
  pack_mask<<<mBlk, 256, 0, stream>>>(msk, mb);

  dim3 gGemm(Csz / 128, Msz / 128);          // (8, 64)
  gemm_wmma<<<gGemm, 256, 0, stream>>>(xq, wqb, bq, Qh, 0);
  gemm_wmma<<<gGemm, 256, 0, stream>>>(xk, wkb, bk, Kh, 0);
  gemm_wmma<<<gGemm, 256, 0, stream>>>(xv, wvb, bv, Vh, 1);

  dim3 gAttn(Tsz / 128, Bsz * Hn);           // (16, 64)
  attn_wmma<<<gAttn, 256, 0, stream>>>(Qh, Kh, Vh, mb, Yb);

  gemm_wmma<<<gGemm, 256, 0, stream>>>(Yb, wfb, bf_, d_out, 2);
}
